// GroundingModule_28080496181375
// MI455X (gfx1250) — matrix operand order repaired and checked by reading, not compiled, not run on hardware
//
#include <hip/hip_runtime.h>
#include <hip/hip_bf16.h>
#include <math.h>

typedef _Float16 f16;
typedef __attribute__((ext_vector_type(16))) _Float16 v16h;
typedef __attribute__((ext_vector_type(8)))  _Float16 v8h;
typedef __attribute__((ext_vector_type(8)))  float    v8f;

static constexpr int DIM  = 768;   // feature dim
static constexpr int SEQ  = 32;    // frames per batch
static constexpr int ASTR = 776;   // LDS row stride in halfs (768 + 8 pad -> 16B-aligned, bank-spread)

__device__ __forceinline__ float gelu_exact(float x) {
    return 0.5f * x * (1.0f + erff(x * 0.7071067811865476f));
}

// A fragment (16x32 f16, MxK): lanes 0-15 -> rows, K = kb+[0..7] & kb+[16..23];
// lanes 16-31 same rows, K = kb+[8..15] & kb+[24..31].  (ISA 7.12.2)
__device__ __forceinline__ v16h load_a_frag(const _Float16* Abuf, int lane, int mb, int kb) {
    int m  = mb + (lane & 15);
    int k0 = kb + 8 * (lane >> 4);
    const _Float16* p = Abuf + m * ASTR + k0;
    v16h r;
    *(v8h*)&r       = *(const v8h*)p;
    *((v8h*)&r + 1) = *(const v8h*)(p + 16);
    return r;
}

// B fragment (32x16 f16, KxN): lane = column; lanes 0-15 hold K=kb+[0..7] & kb+[16..23],
// lanes 16-31 hold K=kb+[8..15] & kb+[24..31]. Wt is [N][DIM] f16, K-contiguous.
__device__ __forceinline__ v16h load_b_frag(const _Float16* Wt, int lane, int nb, int kb) {
    int n  = nb + (lane & 15);
    int k0 = kb + 8 * (lane >> 4);
    const _Float16* p = Wt + (size_t)n * DIM + k0;
    v16h r;
    *(v8h*)&r       = *(const v8h*)p;
    *((v8h*)&r + 1) = *(const v8h*)(p + 16);
    return r;
}

#define WMMA_F16(a, b, c) \
    __builtin_amdgcn_wmma_f32_16x16x32_f16(false, (a), false, (b), (short)0, (c), false, false)

// ---------------------------------------------------------------------------
// Kernel 0: convert + transpose weights to f16 [N][K] layouts (K-contiguous)
// ---------------------------------------------------------------------------
__global__ void prep_weights(const float* __restrict__ vp_w, const float* __restrict__ qp_w,
                             const float* __restrict__ g_w1,
                             f16* __restrict__ WtV, f16* __restrict__ WtQ, f16* __restrict__ WtG1) {
    int i = blockIdx.x * blockDim.x + threadIdx.x;
    if (i < DIM * DIM) {
        int k = i / DIM, n = i % DIM;
        WtV[(size_t)n * DIM + k] = (f16)vp_w[i];
        WtQ[(size_t)n * DIM + k] = (f16)qp_w[i];
    }
    if (i < DIM * (DIM / 2)) {
        int k = i / (DIM / 2), n = i % (DIM / 2);
        WtG1[(size_t)n * DIM + k] = (f16)g_w1[i];
    }
}

// ---------------------------------------------------------------------------
// Kernel 1: qa_p = GELU(LN(qa) @ qp_w + qp_b), 32 rows per block
// ---------------------------------------------------------------------------
__global__ __launch_bounds__(256) void qa_proj_kernel(
    const float* __restrict__ qa, const float* __restrict__ lng, const float* __restrict__ lnb,
    const f16* __restrict__ WtQ, const float* __restrict__ bias, float* __restrict__ qa_p) {
    __shared__ __align__(16) _Float16 Abuf[SEQ * ASTR];
    __shared__ float part1[256], part2[256];
    __shared__ float rowm[SEQ], rowr[SEQ];

    int t = threadIdx.x, lane = t & 31, wave = t >> 5;
    int r = t >> 3, sub = t & 7;
    int row0 = blockIdx.x * SEQ;

    // LayerNorm pass 1: raw -> LDS f16 + stats (8 threads per row)
    const float* src = qa + (size_t)(row0 + r) * DIM;
    float s = 0.f, s2 = 0.f;
    for (int k = 0; k < 96; ++k) {
        int cc = sub + 8 * k;
        float x = src[cc];
        Abuf[r * ASTR + cc] = (f16)x;
        s += x; s2 += x * x;
    }
    part1[t] = s; part2[t] = s2;
    __syncthreads();
    if (t < SEQ) {
        float ss = 0.f, ss2 = 0.f;
        for (int k = 0; k < 8; ++k) { ss += part1[t * 8 + k]; ss2 += part2[t * 8 + k]; }
        float m = ss * (1.0f / 768.f);
        float var = ss2 * (1.0f / 768.f) - m * m;
        rowm[t] = m; rowr[t] = rsqrtf(var + 1e-5f);
    }
    __syncthreads();
    {
        float m = rowm[r], rs = rowr[r];
        for (int k = 0; k < 96; ++k) {
            int cc = sub + 8 * k;
            float x = (float)Abuf[r * ASTR + cc];
            Abuf[r * ASTR + cc] = (f16)(((x - m) * rs) * lng[cc] + lnb[cc]);
        }
    }
    __syncthreads();

    // GEMM: 32x768 @ 768x768; wave handles N slab of 96 (2 M-tiles x 6 N-tiles)
    v8f zacc = {};
    v8f acc[2][6];
    for (int mt = 0; mt < 2; ++mt) for (int nt = 0; nt < 6; ++nt) acc[mt][nt] = zacc;
    int nbase = wave * 96;
    for (int kb = 0; kb < DIM; kb += 32) {
        v16h a0 = load_a_frag(Abuf, lane, 0, kb);
        v16h a1 = load_a_frag(Abuf, lane, 16, kb);
#pragma unroll
        for (int nt = 0; nt < 6; ++nt) {
            v16h bf = load_b_frag(WtQ, lane, nbase + nt * 16, kb);
            acc[0][nt] = WMMA_F16(a0, bf, acc[0][nt]);
            acc[1][nt] = WMMA_F16(a1, bf, acc[1][nt]);
        }
    }

    // epilogue: bias + exact GELU -> qa_p (f32)
    int mhi = (lane >> 4) << 3, nl = lane & 15;
    for (int mt = 0; mt < 2; ++mt)
        for (int nt = 0; nt < 6; ++nt) {
            int n = nbase + nt * 16 + nl;
            float bn = bias[n];
#pragma unroll
            for (int i = 0; i < 8; ++i) {
                int m = mt * 16 + i + mhi;
                qa_p[(size_t)(row0 + m) * DIM + n] = gelu_exact(acc[mt][nt][i] + bn);
            }
        }
}

// ---------------------------------------------------------------------------
// Kernel 2: fully fused per-batch grounding pipeline (one block per batch)
// ---------------------------------------------------------------------------
__global__ __launch_bounds__(256) void grounding_main(
    const float* __restrict__ v,
    const float* __restrict__ vp_lng, const float* __restrict__ vp_lnb,
    const f16* __restrict__ WtV, const float* __restrict__ vp_b,
    const float* __restrict__ g_lng, const float* __restrict__ g_lnb,
    const f16* __restrict__ WtG1, const float* __restrict__ g_b1,
    const float* __restrict__ g_w2, const float* __restrict__ g_b2,
    const float* __restrict__ sigma_p,
    const float* __restrict__ qa_p,
    float* __restrict__ out_keygs, float* __restrict__ out_maxidx,
    float* __restrict__ out_st, float* __restrict__ out_en,
    float* __restrict__ out_mask, float* __restrict__ out_ori) {

    __shared__ __align__(16) _Float16 Abuf[SEQ * ASTR];  // LN(v) -> v_p -> h -> h1(f32)
    __shared__ float qa_sh[DIM];
    __shared__ float part1[256], part2[256];
    __shared__ float rowm[SEQ], rowr[SEQ];
    __shared__ float gatep[SEQ], gate_s[SEQ];
    __shared__ float logits_s[SEQ];

    int t = threadIdx.x, lane = t & 31, wave = t >> 5;
    int r = t >> 3, sub = t & 7;
    int b = blockIdx.x;

    for (int i = t; i < DIM; i += 256) qa_sh[i] = qa_p[(size_t)b * DIM + i];
    if (t < SEQ) gatep[t] = 0.f;

    // ---- LN1 over v rows (v read exactly once from HBM) ----
    const float* src = v + ((size_t)b * SEQ + r) * DIM;
    float s = 0.f, s2 = 0.f;
    for (int k = 0; k < 96; ++k) {
        int cc = sub + 8 * k;
        float x = src[cc];
        Abuf[r * ASTR + cc] = (f16)x;
        s += x; s2 += x * x;
    }
    part1[t] = s; part2[t] = s2;
    __syncthreads();
    if (t < SEQ) {
        float ss = 0.f, ss2 = 0.f;
        for (int k = 0; k < 8; ++k) { ss += part1[t * 8 + k]; ss2 += part2[t * 8 + k]; }
        float m = ss * (1.0f / 768.f);
        float var = ss2 * (1.0f / 768.f) - m * m;
        rowm[t] = m; rowr[t] = rsqrtf(var + 1e-5f);
    }
    __syncthreads();
    {
        float m = rowm[r], rs = rowr[r];
        for (int k = 0; k < 96; ++k) {
            int cc = sub + 8 * k;
            float x = (float)Abuf[r * ASTR + cc];
            Abuf[r * ASTR + cc] = (f16)(((x - m) * rs) * vp_lng[cc] + vp_lnb[cc]);
        }
    }
    __syncthreads();

    // ---- GEMM1: v_p_pre = LN(v) @ vp_w (K=768, N=768) ----
    v8f zacc = {};
    v8f acc[2][6];
    for (int mt = 0; mt < 2; ++mt) for (int nt = 0; nt < 6; ++nt) acc[mt][nt] = zacc;
    int nbase = wave * 96;
    for (int kb = 0; kb < DIM; kb += 32) {
        v16h a0 = load_a_frag(Abuf, lane, 0, kb);
        v16h a1 = load_a_frag(Abuf, lane, 16, kb);
#pragma unroll
        for (int nt = 0; nt < 6; ++nt) {
            v16h bf = load_b_frag(WtV, lane, nbase + nt * 16, kb);
            acc[0][nt] = WMMA_F16(a0, bf, acc[0][nt]);
            acc[1][nt] = WMMA_F16(a1, bf, acc[1][nt]);
        }
    }
    __syncthreads();  // all waves done reading LN(v) from Abuf

    // ---- epilogue1: GELU, v_p -> Abuf (f16), partial gate dots vs qa_p ----
    {
        int mhi = (lane >> 4) << 3, nl = lane & 15;
        float bias_r[6], qv[6];
        for (int nt = 0; nt < 6; ++nt) {
            int n = nbase + nt * 16 + nl;
            bias_r[nt] = vp_b[n];
            qv[nt] = qa_sh[n];
        }
        for (int mt = 0; mt < 2; ++mt)
#pragma unroll
            for (int i = 0; i < 8; ++i) {
                int m = mt * 16 + i + mhi;
                float g = 0.f;
                for (int nt = 0; nt < 6; ++nt) {
                    int n = nbase + nt * 16 + nl;
                    float x = gelu_exact(acc[mt][nt][i] + bias_r[nt]);
                    Abuf[m * ASTR + n] = (f16)x;
                    g += x * qv[nt];
                }
                atomicAdd(&gatep[m], g);
            }
    }
    __syncthreads();
    if (t < SEQ) gate_s[t] = tanhf(gatep[t]);
    __syncthreads();

    // ---- LN2 over v_p * gate (in place, f16 -> f16) ----
    {
        float gl = gate_s[r]; float ss = 0.f, ss2 = 0.f;
        for (int k = 0; k < 96; ++k) {
            int cc = sub + 8 * k;
            float x = (float)Abuf[r * ASTR + cc] * gl;
            ss += x; ss2 += x * x;
        }
        part1[t] = ss; part2[t] = ss2;
    }
    __syncthreads();
    if (t < SEQ) {
        float ss = 0.f, ss2 = 0.f;
        for (int k = 0; k < 8; ++k) { ss += part1[t * 8 + k]; ss2 += part2[t * 8 + k]; }
        float m = ss * (1.0f / 768.f);
        float var = ss2 * (1.0f / 768.f) - m * m;
        rowm[t] = m; rowr[t] = rsqrtf(var + 1e-5f);
    }
    __syncthreads();
    {
        float gl = gate_s[r], m = rowm[r], rs = rowr[r];
        for (int k = 0; k < 96; ++k) {
            int cc = sub + 8 * k;
            float x = (float)Abuf[r * ASTR + cc] * gl;
            Abuf[r * ASTR + cc] = (f16)(((x - m) * rs) * g_lng[cc] + g_lnb[cc]);
        }
    }
    __syncthreads();

    // ---- GEMM2: h @ g_w1 (K=768, N=384); wave N slab of 48 ----
    v8f acc2[2][3];
    for (int mt = 0; mt < 2; ++mt) for (int nt = 0; nt < 3; ++nt) acc2[mt][nt] = zacc;
    int nb2 = wave * 48;
    for (int kb = 0; kb < DIM; kb += 32) {
        v16h a0 = load_a_frag(Abuf, lane, 0, kb);
        v16h a1 = load_a_frag(Abuf, lane, 16, kb);
#pragma unroll
        for (int nt = 0; nt < 3; ++nt) {
            v16h bf = load_b_frag(WtG1, lane, nb2 + nt * 16, kb);
            acc2[0][nt] = WMMA_F16(a0, bf, acc2[0][nt]);
            acc2[1][nt] = WMMA_F16(a1, bf, acc2[1][nt]);
        }
    }
    __syncthreads();  // all waves done reading h

    // ---- epilogue2: GELU -> h1 f32, reuse Abuf region (stride 388 floats) ----
    float* H1 = (float*)Abuf;  // 32*388*4 = 49664 B == Abuf size
    {
        int mhi = (lane >> 4) << 3, nl = lane & 15;
        for (int mt = 0; mt < 2; ++mt)
            for (int nt = 0; nt < 3; ++nt) {
                int n = nb2 + nt * 16 + nl;
                float bn = g_b1[n];
#pragma unroll
                for (int i = 0; i < 8; ++i) {
                    int m = mt * 16 + i + mhi;
                    H1[m * 388 + n] = gelu_exact(acc2[mt][nt][i] + bn);
                }
            }
    }
    __syncthreads();

    // ---- logits = h1 @ g_w2 + b2 ----
    {
        float ss = 0.f;
        for (int k = 0; k < 48; ++k) {
            int cc = sub * 48 + k;
            ss += H1[r * 388 + cc] * g_w2[cc];
        }
        part1[t] = ss;
    }
    __syncthreads();
    if (t < SEQ) {
        float ss = 0.f;
        for (int k = 0; k < 8; ++k) ss += part1[t * 8 + k];
        logits_s[t] = ss + g_b2[0];
    }
    __syncthreads();

    // ---- serial tail (L=32): softmax, gaussian smooth, softmax, interval ----
    if (t == 0) {
        float ori[SEQ], kg[SEQ], cum[SEQ + 1], sm[SEQ];
        float mx = -1e30f;
        for (int l = 0; l < SEQ; ++l) mx = fmaxf(mx, logits_s[l]);
        float sum = 0.f;
        for (int l = 0; l < SEQ; ++l) { float e = expf(logits_s[l] - mx); ori[l] = e; sum += e; }
        float inv = 1.f / sum;
        for (int l = 0; l < SEQ; ++l) ori[l] *= inv;

        float sg = sigma_p[0];
        float kern[5]; float ks = 0.f;
        for (int k = 0; k < 5; ++k) { float xk = (float)(k - 2) / sg; kern[k] = expf(-0.5f * xk * xk); ks += kern[k]; }
        for (int k = 0; k < 5; ++k) kern[k] /= ks;
        for (int l = 0; l < SEQ; ++l) {
            float a = 0.f;
            for (int k = 0; k < 5; ++k) {
                int idx = l + k - 2;
                if (idx >= 0 && idx < SEQ) a += kern[k] * ori[idx];
            }
            sm[l] = a;
        }
        mx = -1e30f;
        for (int l = 0; l < SEQ; ++l) mx = fmaxf(mx, sm[l]);
        sum = 0.f;
        for (int l = 0; l < SEQ; ++l) { float e = expf(sm[l] - mx); kg[l] = e; sum += e; }
        inv = 1.f / sum;
        for (int l = 0; l < SEQ; ++l) kg[l] *= inv;

        int pmax = 0; float bmx = kg[0];
        for (int l = 1; l < SEQ; ++l) if (kg[l] > bmx) { bmx = kg[l]; pmax = l; }
        cum[0] = 0.f;
        for (int l = 0; l < SEQ; ++l) cum[l + 1] = cum[l] + kg[l];
        float best = -1e30f; int st = 0, en = 1;
        const int wlist[3] = {1, 3, 5};
        for (int wi = 0; wi < 3; ++wi) {
            int w = wlist[wi];
            for (int s0 = 0; s0 <= SEQ - w; ++s0) {
                if (pmax >= s0 && pmax < s0 + w) {
                    float sc = cum[s0 + w] - cum[s0];
                    if (sc > best) { best = sc; st = s0; en = s0 + w; }
                }
            }
        }

        float* okg = out_keygs + (size_t)b * SEQ;
        float* oor = out_ori + (size_t)b * SEQ;
        float* oms = out_mask + (size_t)b * SEQ;
        for (int l = 0; l < SEQ; ++l) {
            okg[l] = kg[l];
            oor[l] = ori[l];
            oms[l] = (l >= st && l <= en) ? 1.f : 0.f;
        }
        out_maxidx[(size_t)b * 2 + 0] = (float)st;
        out_maxidx[(size_t)b * 2 + 1] = (float)en;
        out_st[b] = (float)st / 31.f;
        out_en[b] = (float)en / 31.f;
    }
}

// ---------------------------------------------------------------------------
extern "C" void kernel_launch(void* const* d_in, const int* in_sizes, int n_in,
                              void* d_out, int out_size, void* d_ws, size_t ws_size,
                              hipStream_t stream) {
    (void)in_sizes; (void)n_in; (void)out_size; (void)ws_size;
    const float* v      = (const float*)d_in[0];
    const float* qa     = (const float*)d_in[1];
    const float* vp_lng = (const float*)d_in[2];
    const float* vp_lnb = (const float*)d_in[3];
    const float* vp_w   = (const float*)d_in[4];
    const float* vp_b   = (const float*)d_in[5];
    const float* qp_lng = (const float*)d_in[6];
    const float* qp_lnb = (const float*)d_in[7];
    const float* qp_w   = (const float*)d_in[8];
    const float* qp_b   = (const float*)d_in[9];
    const float* g_lng  = (const float*)d_in[10];
    const float* g_lnb  = (const float*)d_in[11];
    const float* g_w1   = (const float*)d_in[12];
    const float* g_b1   = (const float*)d_in[13];
    const float* g_w2   = (const float*)d_in[14];
    const float* g_b2   = (const float*)d_in[15];
    const float* sigma  = (const float*)d_in[16];

    char* ws = (char*)d_ws;
    f16*   WtV  = (f16*)(ws + 0);                 // 768*768*2 = 1179648 B
    f16*   WtQ  = (f16*)(ws + 1179648);           // 1179648 B
    f16*   WtG1 = (f16*)(ws + 2359296);           // 384*768*2 = 589824 B
    float* qa_p = (float*)(ws + 2949120);         // 2048*768*4 = 6291456 B

    float* out = (float*)d_out;
    float* out_keygs  = out;                      // [2048,32]
    float* out_maxidx = out + 65536;              // [2048,2]
    float* out_st     = out + 69632;              // [2048]
    float* out_en     = out + 71680;              // [2048]
    float* out_mask   = out + 73728;              // [2048,32]
    float* out_ori    = out + 139264;             // [2048,32]

    prep_weights<<<(DIM * DIM + 255) / 256, 256, 0, stream>>>(vp_w, qp_w, g_w1, WtV, WtQ, WtG1);
    qa_proj_kernel<<<2048 / SEQ, 256, 0, stream>>>(qa, qp_lng, qp_lnb, WtQ, qp_b, qa_p);
    grounding_main<<<2048, 256, 0, stream>>>(v, vp_lng, vp_lnb, WtV, vp_b,
                                             g_lng, g_lnb, WtG1, g_b1, g_w2, g_b2,
                                             sigma, qa_p,
                                             out_keygs, out_maxidx, out_st, out_en,
                                             out_mask, out_ori);
}
